// MSA_64604898066515
// MI455X (gfx1250) — matrix operand order repaired and checked by reading, not compiled, not run on hardware
//
#include <hip/hip_runtime.h>
#include <hip/hip_bf16.h>

// ---------------------------------------------------------------------------
// MI455X (gfx1250) fused multi-head self-attention, bf16 WMMA pipeline with
// TDM (tensor_load_to_lds) double-buffered staging.
//   B=8, N=1024, W=1024, H=16, D=64
// ---------------------------------------------------------------------------

typedef unsigned short u16;
typedef unsigned int   u32;
typedef __attribute__((ext_vector_type(16))) __bf16 v16bf;
typedef __attribute__((ext_vector_type(8)))  float  v8f;

#if defined(__gfx1250__) && __has_builtin(__builtin_amdgcn_tensor_load_to_lds)
#define USE_TDM 1
#else
#define USE_TDM 0
#endif

__device__ __forceinline__ u16 f2bf(float f) {
    u32 u = __float_as_uint(f);
    u32 r = u + 0x7FFFu + ((u >> 16) & 1u);   // round-to-nearest-even
    return (u16)(r >> 16);
}

#if USE_TDM
typedef __attribute__((ext_vector_type(4))) unsigned int u32x4;
typedef __attribute__((ext_vector_type(8))) int          i32x8;
typedef __attribute__((ext_vector_type(4))) int          i32x4;

// Generic pointers to LDS carry the LDS byte offset in their low 32 bits
// (see ISA 10.2).
__device__ __forceinline__ u32 lds_off32(const void* p) {
    return (u32)(size_t)p;
}

// Issue one TDM 2D tile load: tile d0 x d1 elements (bf16), row stride
// `stride` elements, LDS padding pad_amount (code) dwords every
// 2^(padi+1) dwords.  D# layout per CDNA5 ISA.
// This toolchain exposes the 6-arg builtin:
//   (uint32x4 g0, int32x8 g1, int32x4 g2, int32x4 g3, int32x8 g4, i32 cpol)
__device__ __forceinline__ void tdm_load_2d(u32 lds_byte_off, const u16* gsrc,
                                            u32 d0, u32 d1, u32 stride,
                                            u32 padi, u32 pada) {
    unsigned long long ga = (unsigned long long)(size_t)gsrc;
    u32x4 g0;
    g0[0] = 1u;                                             // count=1, user desc
    g0[1] = lds_byte_off;                                   // lds_addr
    g0[2] = (u32)ga;                                        // global_addr[31:0]
    g0[3] = ((u32)(ga >> 32) & 0x01FFFFFFu) | (2u << 30);   // addr[56:32] | type=2
    i32x8 g1;
    g1[0] = (int)((1u << 16) | (1u << 20) | (padi << 22) | (pada << 25));
             // data_size=2B | pad_enable | pad_interval | pad_amount
    g1[1] = (int)(d0 << 16);   // tensor_dim0[15:0] @ bits 63:48
    g1[2] = (int)(d1 << 16);   // tensor_dim1[15:0] @ bits 95:80
    g1[3] = (int)(d0 << 16);   // tile_dim0        @ bits 127:112
    g1[4] = (int)d1;           // tile_dim1        @ bits 143:128
    g1[5] = (int)stride;       // tensor_dim0_stride[31:0]
    g1[6] = 0;
    g1[7] = 0;
    i32x4 z4 = {0, 0, 0, 0};
    i32x8 z8 = {0, 0, 0, 0, 0, 0, 0, 0};
    __builtin_amdgcn_tensor_load_to_lds(g0, g1, z4, z4, z8, 0);
}
#endif

// ---------------------------------------------------------------------------
// Elementwise fp32 -> bf16 (4 elements / thread)
// ---------------------------------------------------------------------------
__global__ void conv_bf16(const float* __restrict__ src, u16* __restrict__ dst, int n) {
    int i = (blockIdx.x * blockDim.x + threadIdx.x) * 4;
    if (i + 3 < n) {
        float4 v = *(const float4*)(src + i);
        dst[i + 0] = f2bf(v.x);
        dst[i + 1] = f2bf(v.y);
        dst[i + 2] = f2bf(v.z);
        dst[i + 3] = f2bf(v.w);
    }
}

// src [R][C] fp32 -> dst [C][R] bf16 (weights become [out][in]: contiguous-K)
__global__ void transpose_bf16(const float* __restrict__ src, u16* __restrict__ dst,
                               int R, int C) {
    int idx = blockIdx.x * blockDim.x + threadIdx.x;
    if (idx < R * C) {
        int c = idx / R, r = idx % R;
        dst[(size_t)c * R + r] = f2bf(src[(size_t)r * C + c]);
    }
}

// ---------------------------------------------------------------------------
// Tiled bf16 GEMM: C[M,N] = A[M,K] * Bt[N,K]^T, fp32 accumulate via WMMA.
// 128 threads (4 waves); block tile 128x64; each wave 32x64 (2x4 WMMA frags);
// K-step 32; LDS double-buffered, filled by TDM overlapped with compute.
// EPI==1: fp32 row-major output; EPI==2: QKV scatter epilogue.
// ---------------------------------------------------------------------------
template <int EPI>
__global__ __launch_bounds__(128) void gemm_bf16(
    const u16* __restrict__ A, const u16* __restrict__ Bt,
    float* __restrict__ Cf,
    u16* __restrict__ Qo, u16* __restrict__ Ko, u16* __restrict__ Vto,
    int M, int N, int K)
{
    __shared__ u16 As[2][128 * 40];
    __shared__ u16 Bs[2][64 * 40];

    const int tid  = threadIdx.x;
    const int wave = tid >> 5;
    const int lane = tid & 31;
    const int l16  = lane & 15;
    const int lhi  = lane >> 4;
    const int row0 = blockIdx.x * 128;
    const int col0 = blockIdx.y * 64;
    const int nk   = K >> 5;

    v8f acc[2][4];
#pragma unroll
    for (int rf = 0; rf < 2; rf++)
#pragma unroll
        for (int nf = 0; nf < 4; nf++)
#pragma unroll
            for (int j = 0; j < 8; j++) acc[rf][nf][j] = 0.0f;

#if USE_TDM
    if (wave == 0) {
        tdm_load_2d(lds_off32(&As[0][0]), A + (size_t)row0 * K, 32, 128, (u32)K, 3, 3);
        tdm_load_2d(lds_off32(&Bs[0][0]), Bt + (size_t)col0 * K, 32, 64, (u32)K, 3, 3);
    }
#endif

    for (int t = 0; t < nk; t++) {
        const int cur = t & 1;
#if USE_TDM
        if (wave == 0) __builtin_amdgcn_s_wait_tensorcnt(0);
        __syncthreads();
        if (wave == 0 && t + 1 < nk) {
            tdm_load_2d(lds_off32(&As[cur ^ 1][0]),
                        A + (size_t)row0 * K + (t + 1) * 32, 32, 128, (u32)K, 3, 3);
            tdm_load_2d(lds_off32(&Bs[cur ^ 1][0]),
                        Bt + (size_t)col0 * K + (t + 1) * 32, 32, 64, (u32)K, 3, 3);
        }
#else
        __syncthreads();
        {   // cooperative staging fallback
            int k0 = t * 32;
#pragma unroll
            for (int i = tid; i < 256; i += 128) {
                int r = i >> 1, h = i & 1;
                *(float4*)(&As[cur][r * 40 + h * 16]) =
                    *(const float4*)(A + (size_t)(row0 + r) * K + k0 + h * 16);
            }
            {
                int r = tid >> 1, h = tid & 1;
                *(float4*)(&Bs[cur][r * 40 + h * 16]) =
                    *(const float4*)(Bt + (size_t)(col0 + r) * K + k0 + h * 16);
            }
        }
        __syncthreads();
#endif
        const u16* as = As[cur];
        const u16* bs = Bs[cur];

        union Fr { v16bf v; u32 u[8]; };
        Fr a[2];
#pragma unroll
        for (int rf = 0; rf < 2; rf++)
#pragma unroll
            for (int p = 0; p < 8; p++) {
                int kk = ((p >> 2) << 4) + (lhi << 3) + ((p & 3) << 1);
                a[rf].u[p] = *(const u32*)(&as[(wave * 32 + rf * 16 + l16) * 40 + kk]);
            }
#pragma unroll
        for (int nf = 0; nf < 4; nf++) {
            Fr b;
#pragma unroll
            for (int p = 0; p < 8; p++) {
                int kk = ((p >> 2) << 4) + (lhi << 3) + ((p & 3) << 1);
                b.u[p] = *(const u32*)(&bs[(nf * 16 + l16) * 40 + kk]);
            }
#pragma unroll
            for (int rf = 0; rf < 2; rf++)
                acc[rf][nf] = __builtin_amdgcn_wmma_f32_16x16x32_bf16(
                    false, a[rf].v, false, b.v, (short)0, acc[rf][nf], false, false);
        }
    }

#pragma unroll
    for (int rf = 0; rf < 2; rf++)
#pragma unroll
        for (int nf = 0; nf < 4; nf++)
#pragma unroll
            for (int j = 0; j < 8; j++) {
                int gr = row0 + wave * 32 + rf * 16 + j + (lhi << 3);
                int gc = col0 + nf * 16 + l16;
                float v = acc[rf][nf][j];
                if (EPI == 1) {
                    Cf[(size_t)gr * N + gc] = v;
                } else {   // QKV scatter
                    int bb  = gr >> 10, n = gr & 1023;
                    int sec = gc >> 10, wo = gc & 1023;
                    int h   = wo >> 6,  d = wo & 63;
                    size_t bh = (size_t)(bb * 16 + h);
                    if (sec == 0)      Qo [(bh * 1024 + n) * 64 + d] = f2bf(v * 0.125f);
                    else if (sec == 1) Ko [(bh * 1024 + n) * 64 + d] = f2bf(v);
                    else               Vto[(bh * 64 + d) * 1024 + n] = f2bf(v);
                }
            }
}

// ---------------------------------------------------------------------------
// Flash-style attention: block = (bh, 64-query tile); 4 waves x 16 rows.
// K/V tiles TDM double-buffered; online softmax with shfl_xor reductions;
// S->P relayout through LDS for the PV WMMA.
// ---------------------------------------------------------------------------
__global__ __launch_bounds__(128) void attn_kernel(
    const u16* __restrict__ Q, const u16* __restrict__ Kd,
    const u16* __restrict__ Vt, u16* __restrict__ Ao)
{
    __shared__ u16 Ks[2][64 * 72];
    __shared__ u16 Vs[2][64 * 72];
    __shared__ u16 Ps[4 * 16 * 72];

    const int tid  = threadIdx.x;
    const int wave = tid >> 5;
    const int lane = tid & 31;
    const int l16  = lane & 15;
    const int lhi  = lane >> 4;
    const int bh   = blockIdx.y;                // b*16 + h
    const int q0   = blockIdx.x * 64;
    const size_t base = (size_t)bh * 1024 * 64;

    union F { v16bf v; u32 u[8]; };

    // preload Q fragments (16 rows x 64 D -> 2 K-steps), Q pre-scaled 1/sqrt(D)
    F aQ[2];
#pragma unroll
    for (int ks = 0; ks < 2; ks++)
#pragma unroll
        for (int p = 0; p < 8; p++) {
            int kk = ks * 32 + ((p >> 2) << 4) + (lhi << 3) + ((p & 3) << 1);
            int r  = q0 + wave * 16 + l16;
            aQ[ks].u[p] = *(const u32*)(Q + base + (size_t)r * 64 + kk);
        }

    v8f  o[4];
    float m[8], l[8];
#pragma unroll
    for (int i = 0; i < 4; i++)
#pragma unroll
        for (int j = 0; j < 8; j++) o[i][j] = 0.0f;
#pragma unroll
    for (int j = 0; j < 8; j++) { m[j] = -1e30f; l[j] = 0.0f; }

#if USE_TDM
    if (wave == 0) {
        tdm_load_2d(lds_off32(&Ks[0][0]), Kd + base, 64, 64, 64, 4, 3);
        tdm_load_2d(lds_off32(&Vs[0][0]), Vt + base, 64, 64, 1024, 4, 3);
    }
#endif

    for (int kt = 0; kt < 16; kt++) {
        const int cur = kt & 1;
#if USE_TDM
        if (wave == 0) __builtin_amdgcn_s_wait_tensorcnt(0);
        __syncthreads();
        if (wave == 0 && kt + 1 < 16) {
            tdm_load_2d(lds_off32(&Ks[cur ^ 1][0]),
                        Kd + base + (size_t)(kt + 1) * 64 * 64, 64, 64, 64, 4, 3);
            tdm_load_2d(lds_off32(&Vs[cur ^ 1][0]),
                        Vt + base + (kt + 1) * 64, 64, 64, 1024, 4, 3);
        }
#else
        __syncthreads();
        for (int i = tid; i < 256; i += 128) {
            int r = i >> 2, s = i & 3;
            *(float4*)(&Ks[cur][r * 72 + s * 16]) =
                *(const float4*)(Kd + base + (size_t)(kt * 64 + r) * 64 + s * 16);
            *(float4*)(&Vs[cur][r * 72 + s * 16]) =
                *(const float4*)(Vt + base + (size_t)r * 1024 + kt * 64 + s * 16);
        }
        __syncthreads();
#endif
        const u16* ks_t = Ks[cur];
        const u16* vs_t = Vs[cur];

        // S = Q * K^T
        v8f s[4];
#pragma unroll
        for (int nf = 0; nf < 4; nf++) {
#pragma unroll
            for (int j = 0; j < 8; j++) s[nf][j] = 0.0f;
#pragma unroll
            for (int ks = 0; ks < 2; ks++) {
                F b;
#pragma unroll
                for (int p = 0; p < 8; p++) {
                    int kk = ks * 32 + ((p >> 2) << 4) + (lhi << 3) + ((p & 3) << 1);
                    b.u[p] = *(const u32*)(&ks_t[(nf * 16 + l16) * 72 + kk]);
                }
                s[nf] = __builtin_amdgcn_wmma_f32_16x16x32_bf16(
                    false, aQ[ks].v, false, b.v, (short)0, s[nf], false, false);
            }
        }

        // online softmax: row = j + 8*(lane/16), reduce across 16-lane group
#pragma unroll
        for (int j = 0; j < 8; j++) {
            float mx = s[0][j];
#pragma unroll
            for (int nf = 1; nf < 4; nf++) mx = fmaxf(mx, s[nf][j]);
#pragma unroll
            for (int off = 1; off < 16; off <<= 1)
                mx = fmaxf(mx, __shfl_xor(mx, off, 32));
            float nm = fmaxf(m[j], mx);
            float sc = __expf(m[j] - nm);
            float rs = 0.0f;
#pragma unroll
            for (int nf = 0; nf < 4; nf++) {
                float p = __expf(s[nf][j] - nm);
                s[nf][j] = p;
                rs += p;
            }
#pragma unroll
            for (int off = 1; off < 16; off <<= 1)
                rs += __shfl_xor(rs, off, 32);
            l[j] = l[j] * sc + rs;
            m[j] = nm;
#pragma unroll
            for (int nf = 0; nf < 4; nf++) o[nf][j] *= sc;
        }

        // relayout P (C-layout f32) -> bf16 A-fragments via per-wave LDS region
#pragma unroll
        for (int nf = 0; nf < 4; nf++)
#pragma unroll
            for (int j = 0; j < 8; j++)
                Ps[(wave * 16 + j + (lhi << 3)) * 72 + nf * 16 + l16] = f2bf(s[nf][j]);
        __syncthreads();

        F aP[2];
#pragma unroll
        for (int ks = 0; ks < 2; ks++)
#pragma unroll
            for (int p = 0; p < 8; p++) {
                int kk = ks * 32 + ((p >> 2) << 4) + (lhi << 3) + ((p & 3) << 1);
                aP[ks].u[p] = *(const u32*)(&Ps[(wave * 16 + l16) * 72 + kk]);
            }

        // O += P * V  (B frag from V^T tile: n = d, k = key)
#pragma unroll
        for (int nf = 0; nf < 4; nf++) {
#pragma unroll
            for (int ks = 0; ks < 2; ks++) {
                F b;
#pragma unroll
                for (int p = 0; p < 8; p++) {
                    int kk = ks * 32 + ((p >> 2) << 4) + (lhi << 3) + ((p & 3) << 1);
                    b.u[p] = *(const u32*)(&vs_t[(nf * 16 + l16) * 72 + kk]);
                }
                o[nf] = __builtin_amdgcn_wmma_f32_16x16x32_bf16(
                    false, aP[ks].v, false, b.v, (short)0, o[nf], false, false);
            }
        }
    }

    // normalize and store attn output token-major [B,N,W] as bf16
    const int bb = bh >> 4, h = bh & 15;
#pragma unroll
    for (int nf = 0; nf < 4; nf++)
#pragma unroll
        for (int j = 0; j < 8; j++) {
            int n = q0 + wave * 16 + j + (lhi << 3);
            int d = nf * 16 + l16;
            float v = o[nf][j] / l[j];
            Ao[((size_t)(bb * 1024 + n)) * 1024 + h * 64 + d] = f2bf(v);
        }
}

// ---------------------------------------------------------------------------
// Launch
// ---------------------------------------------------------------------------
extern "C" void kernel_launch(void* const* d_in, const int* in_sizes, int n_in,
                              void* d_out, int out_size, void* d_ws, size_t ws_size,
                              hipStream_t stream) {
    const float* x     = (const float*)d_in[0];   // [8,1024,1024]
    const float* w_qkv = (const float*)d_in[1];   // [1024,3072]
    const float* w_out = (const float*)d_in[2];   // [1024,1024]
    float*       out   = (float*)d_out;           // [8,1024,1024]

    char* ws = (char*)d_ws;
    const size_t MB = 1024 * 1024;
    u16* xb   = (u16*)(ws + 0 * MB);   // x bf16            16 MB
    u16* wqkt = (u16*)(ws + 16 * MB);  // w_qkv^T bf16       6 MB
    u16* wot  = (u16*)(ws + 22 * MB);  // w_out^T bf16       2 MB
    u16* Qb   = (u16*)(ws + 24 * MB);  // Q [B,H,N,D]       16 MB
    u16* Kb   = (u16*)(ws + 40 * MB);  // K [B,H,N,D]       16 MB
    u16* Vtb  = (u16*)(ws + 56 * MB);  // V^T [B,H,D,N]     16 MB
    u16* Ab   = (u16*)(ws + 72 * MB);  // attn out [B,N,W]  16 MB

    conv_bf16<<<8192, 256, 0, stream>>>(x, xb, 8 * 1024 * 1024);
    transpose_bf16<<<(3 * 1024 * 1024 + 255) / 256, 256, 0, stream>>>(w_qkv, wqkt, 1024, 3072);
    transpose_bf16<<<(1024 * 1024 + 255) / 256, 256, 0, stream>>>(w_out, wot, 1024, 1024);

    // QKV projection: [8192,1024] x [1024,3072] with head-major scatter
    gemm_bf16<2><<<dim3(64, 48), 128, 0, stream>>>(
        xb, wqkt, nullptr, Qb, Kb, Vtb, 8192, 3072, 1024);

    // attention: 128 (b,h) pairs x 16 query tiles
    attn_kernel<<<dim3(16, 128), 128, 0, stream>>>(Qb, Kb, Vtb, Ab);

    // output projection -> fp32
    gemm_bf16<1><<<dim3(64, 16), 128, 0, stream>>>(
        Ab, wot, out, nullptr, nullptr, nullptr, 8192, 1024, 1024);
}
